// SequenceTaggingModel_16578573762740
// MI455X (gfx1250) — matrix operand order repaired and checked by reading, not compiled, not run on hardware
//
#include <hip/hip_runtime.h>
#include <stdint.h>

// Problem dims
#define LDIM 256   // sequence length
#define BDIM 256   // batch
#define EDIM 300   // embed dim
#define KX   320   // padded embed dim (mult of 32)
#define HDIM 256   // hidden
#define G4   1024  // 4*H
#define TT   17    // tags
#define TP   32    // padded tags
#define NWG  16    // workgroups per LSTM direction

#define LDS_BYTES (64 * KX * 2 + 64 * HDIM * 2)   // 73728 B per WG

typedef __attribute__((ext_vector_type(16))) __bf16 bf16x16;
typedef __attribute__((ext_vector_type(8)))  float  f32x8;

union FragU { uint4 u[2]; bf16x16 v; };
union C8   { float4 q[2]; float f[8]; };

__device__ __forceinline__ uint16_t f2bf(float f) {
  uint32_t u = __float_as_uint(f);
  u += 0x7FFFu + ((u >> 16) & 1u);   // round-to-nearest-even
  return (uint16_t)(u >> 16);
}

// A fragment (16x32 bf16, M x K): lane holds row m=lane%16;
// lanes 0-15: K kb..kb+15; lanes 16-31: K kb+16..kb+31
__device__ __forceinline__ bf16x16 ldA(const uint16_t* rowbase, int kb, int half) {
  const uint16_t* p = rowbase + kb + half * 16;
  FragU f;
  f.u[0] = *(const uint4*)(p);
  f.u[1] = *(const uint4*)(p + 8);
  return f.v;
}
// B fragment (32x16 bf16, K x N) from global memory: lane holds column
// n=lane%16 (== row n of the row-major weight); lanes 0-15: K kb..kb+15,
// lanes 16-31: K kb+16..kb+31
__device__ __forceinline__ bf16x16 ldB(const uint16_t* rowbase, int kb, int half) {
  const uint16_t* p = rowbase + kb + half * 16;
  FragU f;
  f.u[0] = *(const uint4*)(p);
  f.u[1] = *(const uint4*)(p + 8);
  return f.v;
}
// B fragment from LDS via volatile asm ds_load_b128: cannot be hoisted,
// batched or spilled by the compiler — lands 2 instructions before its WMMA.
// Explicit s_wait_dscnt is mandatory (compiler can't track asm-issued loads).
__device__ __forceinline__ bf16x16 ldB_lds(const uint16_t* rowbase, int kb, int half) {
  uint32_t off = (uint32_t)(uintptr_t)(rowbase + kb + half * 16);
  FragU f;
  asm volatile("ds_load_b128 %0, %2\n\t"
               "ds_load_b128 %1, %2 offset:16\n\t"
               "s_wait_dscnt 0x0"
               : "=v"(f.u[0]), "=v"(f.u[1])
               : "v"(off));
  return f.v;
}
__device__ __forceinline__ f32x8 wmma_bf16(bf16x16 a, bf16x16 b, f32x8 c) {
  return __builtin_amdgcn_wmma_f32_16x16x32_bf16(false, a, false, b, (short)0, c, false, false);
}
__device__ __forceinline__ float sigm(float x) { return 1.0f / (1.0f + __expf(-x)); }

// ---------------------------------------------------------------- init (zero)
__global__ void k_init(uint32_t* p, int nwords) {
  for (int i = blockIdx.x * blockDim.x + threadIdx.x; i < nwords;
       i += gridDim.x * blockDim.x)
    p[i] = 0u;
}

// ------------------------------------------------- weight prep (f32 -> bf16)
__global__ void k_prep(const float* wihf, const float* whhf,
                       const float* bihf, const float* bhhf,
                       const float* wihb, const float* whhb,
                       const float* bihb, const float* bhhb,
                       const float* linw_f32,
                       uint16_t* Wih16, uint16_t* Whh16, float* bias,
                       uint16_t* linw16) {
  const int n0 = 2 * G4 * KX;        // padded W_ih, both dirs
  const int n1 = 2 * G4 * HDIM;      // W_hh, both dirs
  const int n2 = 2 * G4;             // combined bias
  const int n3 = TP * (2 * HDIM);    // padded lin_w
  const int total = n0 + n1 + n2 + n3;
  for (int i = blockIdx.x * blockDim.x + threadIdx.x; i < total;
       i += gridDim.x * blockDim.x) {
    if (i < n0) {
      int d = i / (G4 * KX), r = (i / KX) % G4, k = i % KX;
      const float* w = d ? wihb : wihf;
      Wih16[i] = (k < EDIM) ? f2bf(w[(size_t)r * EDIM + k]) : (uint16_t)0;
    } else if (i < n0 + n1) {
      int j = i - n0;
      int d = j / (G4 * HDIM), r = (j / HDIM) % G4, k = j % HDIM;
      const float* w = d ? whhb : whhf;
      Whh16[j] = f2bf(w[(size_t)r * HDIM + k]);
    } else if (i < n0 + n1 + n2) {
      int j = i - n0 - n1;
      int d = j / G4, r = j % G4;
      bias[j] = d ? (bihb[r] + bhhb[r]) : (bihf[r] + bhhf[r]);
    } else {
      int j = i - n0 - n1 - n2;
      int n = j / (2 * HDIM), k = j % (2 * HDIM);
      linw16[j] = (n < TT) ? f2bf(linw_f32[(size_t)n * (2 * HDIM) + k]) : (uint16_t)0;
    }
  }
}

// ------------------------------------------ embedding gather (f32 -> bf16 X)
__global__ void k_embed(const int* words, const float* embed, uint16_t* X16) {
  const int total = LDIM * BDIM * KX;
  for (int i = blockIdx.x * blockDim.x + threadIdx.x; i < total;
       i += gridDim.x * blockDim.x) {
    int row = i / KX, k = i % KX;
    int w = words[row];
    X16[i] = (k < EDIM) ? f2bf(embed[(size_t)w * EDIM + k]) : (uint16_t)0;
  }
}

// ------------------------------------------- persistent bidirectional LSTM
// grid = (NWG, 2): blockIdx.y = direction. 512 thr = 16 waves per WG.
// WG bx owns h-tile h0 = bx*16; wave wv owns b-tile b0 = wv*16 — exactly one
// 16x16 output tile per wave, 256 tiles per direction. Per-WG weight slice
// (64 rows W_ih + 64 rows W_hh = 72 KB) staged to LDS with async
// global->LDS b128 copies; per-step B fragments come from asm ds_load_b128.
__global__ void __launch_bounds__(512, 1)
k_lstm(const uint16_t* __restrict__ X16, const uint16_t* __restrict__ Wih16,
       const uint16_t* __restrict__ Whh16, const float* __restrict__ bias,
       uint16_t* hbuf, float* cws, uint16_t* lstm, int* bar) {
  extern __shared__ uint16_t smem[];
  uint16_t* sWih = smem;              // [64][KX]   rows: g*16 + c
  uint16_t* sWhh = smem + 64 * KX;    // [64][HDIM] rows: g*16 + c

  const int lane = threadIdx.x & 31;
  const int half = lane >> 4;
  const int l16  = lane & 15;
  const int wv   = threadIdx.x >> 5;
  const int dir  = blockIdx.y;
  const int bx   = blockIdx.x;

  const uint16_t* WihD  = Wih16 + (size_t)dir * G4 * KX;
  const uint16_t* WhhD  = Whh16 + (size_t)dir * G4 * HDIM;
  const float*    biasD = bias + dir * G4;
  uint16_t* hb = hbuf + (size_t)dir * 2 * BDIM * HDIM;   // double buffer
  float*    cD = cws + (size_t)dir * 256 * 32 * 8;
  int* cnt = bar + dir * 32;
  int* gen = bar + dir * 32 + 16;

  const int h0 = bx << 4;     // this WG's h-tile
  const int b0 = wv << 4;     // this wave's b-tile
  const int tp = bx * 16 + wv;

  // ---- stage this WG's weight slice into LDS (async global->LDS b128) ----
  // local row lr = g*16 + c  <->  global W row g*HDIM + h0 + c
  {
    const int NQ_IH = 64 * (KX / 8);     // 2560 x 16B
    const int NQ_HH = 64 * (HDIM / 8);   // 2048 x 16B
    for (int i = threadIdx.x; i < NQ_IH + NQ_HH; i += 512) {
      const uint16_t* gp;
      uint16_t* lp;
      if (i < NQ_IH) {
        int lr = i / (KX / 8), q = i % (KX / 8);
        int g = lr >> 4, c = lr & 15;
        gp = WihD + (size_t)(g * HDIM + h0 + c) * KX + q * 8;
        lp = sWih + (size_t)lr * KX + q * 8;
      } else {
        int j = i - NQ_IH;
        int lr = j / (HDIM / 8), q = j % (HDIM / 8);
        int g = lr >> 4, c = lr & 15;
        gp = WhhD + (size_t)(g * HDIM + h0 + c) * HDIM + q * 8;
        lp = sWhh + (size_t)lr * HDIM + q * 8;
      }
      uint32_t loff = (uint32_t)(uintptr_t)lp;   // low 32 bits == LDS byte offset
      asm volatile("global_load_async_to_lds_b128 %0, %1, off"
                   :: "v"(loff), "v"(gp) : "memory");
    }
    asm volatile("s_wait_asynccnt 0x0" ::: "memory");
    __syncthreads();
  }

  const f32x8 vzero = {0.f, 0.f, 0.f, 0.f, 0.f, 0.f, 0.f, 0.f};

  // per-gate LDS row bases (rows g*16 + l16)
  const uint16_t* wihRow[4];
  const uint16_t* whhRow[4];
  #pragma unroll
  for (int g = 0; g < 4; ++g) {
    wihRow[g] = sWih + (size_t)(g * 16 + l16) * KX;
    whhRow[g] = sWhh + (size_t)(g * 16 + l16) * HDIM;
  }
  const float bI = biasD[0 * HDIM + h0 + l16];
  const float bF = biasD[1 * HDIM + h0 + l16];
  const float bG = biasD[2 * HDIM + h0 + l16];
  const float bO = biasD[3 * HDIM + h0 + l16];
  float* cp = cD + ((size_t)tp * 32 + lane) * 8;

  for (int s = 0; s < LDIM; ++s) {
    const int l = dir ? (LDIM - 1 - s) : s;
    const uint16_t* hprev = hb + (size_t)(s & 1) * BDIM * HDIM;
    uint16_t*       hnext = hb + (size_t)((s + 1) & 1) * BDIM * HDIM;

    f32x8 acc[4];
    acc[0] = vzero; acc[1] = vzero; acc[2] = vzero; acc[3] = vzero;

    // input projection part: K = 320 (padded E), weights from LDS
    const uint16_t* Xrow = X16 + ((size_t)(l * BDIM + b0 + l16)) * KX;
    for (int kb = 0; kb < KX; kb += 32) {
      bf16x16 a = ldA(Xrow, kb, half);
      #pragma unroll
      for (int g = 0; g < 4; ++g)
        acc[g] = wmma_bf16(a, ldB_lds(wihRow[g], kb, half), acc[g]);
    }
    // recurrent part: K = 256, weights from LDS
    const uint16_t* hrow = hprev + (size_t)(b0 + l16) * HDIM;
    for (int kb = 0; kb < HDIM; kb += 32) {
      bf16x16 a = ldA(hrow, kb, half);
      #pragma unroll
      for (int g = 0; g < 4; ++g)
        acc[g] = wmma_bf16(a, ldB_lds(whhRow[g], kb, half), acc[g]);
    }

    C8 cold, cnew;
    cold.q[0] = *(float4*)cp;
    cold.q[1] = *(float4*)(cp + 4);

    #pragma unroll
    for (int r = 0; r < 8; ++r) {
      float iv = sigm(acc[0][r] + bI);
      float fv = sigm(acc[1][r] + bF);
      float gv = tanhf(acc[2][r] + bG);
      float ov = sigm(acc[3][r] + bO);
      float cc = fv * cold.f[r] + iv * gv;
      float hv = ov * tanhf(cc);
      cnew.f[r] = cc;
      uint16_t h16 = f2bf(hv);
      int brow = b0 + r + half * 8;                    // C layout: m=r+8*half
      hnext[(size_t)brow * HDIM + h0 + l16] = h16;
      lstm[((size_t)l * BDIM + brow) * (2 * HDIM) + dir * HDIM + h0 + l16] = h16;
    }
    *(float4*)cp = cnew.q[0];
    *(float4*)(cp + 4) = cnew.q[1];

    // device-scope per-direction barrier (16 WGs)
    __threadfence();
    __syncthreads();
    if (threadIdx.x == 0) {
      if (atomicAdd(cnt, 1) == NWG - 1) {
        atomicExch(cnt, 0);
        atomicAdd(gen, 1);
      } else {
        while (__hip_atomic_load(gen, __ATOMIC_RELAXED, __HIP_MEMORY_SCOPE_AGENT) < s + 1)
          __builtin_amdgcn_s_sleep(2);
      }
    }
    __syncthreads();
    __threadfence();
  }
}

// -------------------------------------------- emissions: lstm_out @ lin_w^T
// 65536x512 (bf16) x 512x32 (bf16, tags padded) -> 65536x32 f32
__global__ void __launch_bounds__(256)
k_emis(const uint16_t* __restrict__ lstm, const uint16_t* __restrict__ linw16,
       const float* __restrict__ linb, float* __restrict__ emis) {
  const int lane = threadIdx.x & 31;
  const int half = lane >> 4;
  const int l16  = lane & 15;
  const int tile = blockIdx.x * 8 + (threadIdx.x >> 5);  // 8192 tiles
  const int mt = tile >> 1, nt = tile & 1;

  const uint16_t* arow = lstm + ((size_t)mt * 16 + l16) * (2 * HDIM);
  const uint16_t* brow = linw16 + ((size_t)(nt * 16 + l16)) * (2 * HDIM);
  f32x8 acc = {0.f, 0.f, 0.f, 0.f, 0.f, 0.f, 0.f, 0.f};
  #pragma unroll
  for (int kc = 0; kc < 2 * HDIM; kc += 32)
    acc = wmma_bf16(ldA(arow, kc, half), ldB(brow, kc, half), acc);

  const int col = nt * 16 + l16;
  const float lb = (col < TT) ? linb[col] : 0.f;
  #pragma unroll
  for (int r = 0; r < 8; ++r) {
    int row = mt * 16 + r + half * 8;
    emis[(size_t)row * TP + col] = acc[r] + lb;
  }
}

// ---------------------------------- CRF: gold score + forward logZ + reduce
__global__ void __launch_bounds__(256)
k_crf(const float* __restrict__ emis, const int* __restrict__ tags,
      const int* __restrict__ mask, const float* __restrict__ startT,
      const float* __restrict__ endT, const float* __restrict__ transT,
      float* out) {
  __shared__ float tr[TT][TT];
  __shared__ float st[TT], en[TT];
  __shared__ float red[BDIM];
  const int b = threadIdx.x;
  if (b < TT * TT) tr[b / TT][b % TT] = transT[b];
  if (b < TT) { st[b] = startT[b]; en[b] = endT[b]; }
  __syncthreads();

  float alpha[TT];
  const float* em0 = emis + (size_t)b * TP;
  #pragma unroll
  for (int j = 0; j < TT; ++j) alpha[j] = st[j] + em0[j];
  int prev = tags[b];
  float score = alpha[prev];  // start[t0] + em0[t0]
  int se = mask[b];

  for (int l = 1; l < LDIM; ++l) {
    const float* eml = emis + ((size_t)l * BDIM + b) * TP;
    const int tg = tags[l * BDIM + b];
    const int m  = mask[l * BDIM + b];
    score += (float)m * (tr[prev][tg] + eml[tg]);
    prev = tg;
    se += m;

    float nxt[TT];
    #pragma unroll
    for (int j = 0; j < TT; ++j) {
      float mx = -1e30f;
      #pragma unroll
      for (int i = 0; i < TT; ++i) mx = fmaxf(mx, alpha[i] + tr[i][j]);
      float sm = 0.f;
      #pragma unroll
      for (int i = 0; i < TT; ++i) sm += __expf(alpha[i] + tr[i][j] - mx);
      nxt[j] = mx + __logf(sm) + eml[j];
    }
    #pragma unroll
    for (int j = 0; j < TT; ++j) alpha[j] = m ? nxt[j] : alpha[j];
  }

  score += en[tags[(se - 1) * BDIM + b]];
  float mx = -1e30f;
  #pragma unroll
  for (int j = 0; j < TT; ++j) mx = fmaxf(mx, alpha[j] + en[j]);
  float sm = 0.f;
  #pragma unroll
  for (int j = 0; j < TT; ++j) sm += __expf(alpha[j] + en[j] - mx);
  const float logZ = mx + __logf(sm);

  red[b] = score - logZ;
  __syncthreads();
  for (int off = 128; off > 0; off >>= 1) {
    if (b < off) red[b] += red[b + off];
    __syncthreads();
  }
  if (b == 0) out[0] = -red[0];
}

// --------------------------------------------------------------------- host
extern "C" void kernel_launch(void* const* d_in, const int* in_sizes, int n_in,
                              void* d_out, int out_size, void* d_ws, size_t ws_size,
                              hipStream_t stream) {
  (void)in_sizes; (void)n_in; (void)out_size; (void)ws_size;

  const int*   words = (const int*)d_in[0];
  const int*   tags  = (const int*)d_in[1];
  const int*   mask  = (const int*)d_in[2];
  const float* embed = (const float*)d_in[3];
  const float* wihf  = (const float*)d_in[4];
  const float* whhf  = (const float*)d_in[5];
  const float* bihf  = (const float*)d_in[6];
  const float* bhhf  = (const float*)d_in[7];
  const float* wihb  = (const float*)d_in[8];
  const float* whhb  = (const float*)d_in[9];
  const float* bihb  = (const float*)d_in[10];
  const float* bhhb  = (const float*)d_in[11];
  const float* linw  = (const float*)d_in[12];
  const float* linb  = (const float*)d_in[13];
  const float* stT   = (const float*)d_in[14];
  const float* enT   = (const float*)d_in[15];
  const float* trT   = (const float*)d_in[16];

  char* w = (char*)d_ws;
  size_t o = 0;
  auto take = [&](size_t bytes) -> char* {
    char* p = w + o;
    o += (bytes + 255) & ~(size_t)255;
    return p;
  };
  int*      bar    = (int*)take(256);                                  // barriers
  float*    cws    = (float*)take((size_t)2 * 256 * 32 * 8 * 4);       // c state
  uint16_t* hbuf   = (uint16_t*)take((size_t)2 * 2 * BDIM * HDIM * 2); // h dbl-buf
  uint16_t* Wih16  = (uint16_t*)take((size_t)2 * G4 * KX * 2);
  uint16_t* Whh16  = (uint16_t*)take((size_t)2 * G4 * HDIM * 2);
  float*    biasWS = (float*)take((size_t)2 * G4 * 4);
  uint16_t* linw16 = (uint16_t*)take((size_t)TP * 2 * HDIM * 2);
  uint16_t* X16    = (uint16_t*)take((size_t)LDIM * BDIM * KX * 2);
  uint16_t* lstm   = (uint16_t*)take((size_t)LDIM * BDIM * 2 * HDIM * 2);
  float*    emis   = (float*)take((size_t)LDIM * BDIM * TP * 4);

  // zero bar + c + h (they are the first three contiguous segments)
  const int zwords = (int)((256 + (size_t)2 * 256 * 32 * 8 * 4 +
                            (size_t)2 * 2 * BDIM * HDIM * 2) / 4);
  k_init<<<(zwords + 255) / 256, 256, 0, stream>>>((uint32_t*)d_ws, zwords);

  const int prepN = 2 * G4 * KX + 2 * G4 * HDIM + 2 * G4 + TP * 2 * HDIM;
  k_prep<<<(prepN + 255) / 256, 256, 0, stream>>>(
      wihf, whhf, bihf, bhhf, wihb, whhb, bihb, bhhb, linw,
      Wih16, Whh16, biasWS, linw16);

  k_embed<<<(LDIM * BDIM * KX + 255) / 256, 256, 0, stream>>>(words, embed, X16);

  k_lstm<<<dim3(NWG, 2), 512, LDS_BYTES, stream>>>(X16, Wih16, Whh16, biasWS,
                                                   hbuf, cws, lstm, bar);

  k_emis<<<(LDIM * BDIM / 16) * 2 / 8, 256, 0, stream>>>(lstm, linw16, linb, emis);

  k_crf<<<1, 256, 0, stream>>>(emis, tags, mask, stT, enT, trT, (float*)d_out);
}
